// Reservoir2D_61057255080268
// MI455X (gfx1250) — matrix operand order repaired and checked by reading, not compiled, not run on hardware
//
#include <hip/hip_runtime.h>
#include <hip/hip_bf16.h>

// ---------------- problem constants ----------------
#define BATCH   64
#define NXg     32
#define NYg     32
#define NCELL   1024          // NXg*NYg
#define NSTEP   1000
#define LEN_Y   50
#define LEN_T   20            // NSTEP/LEN_Y
#define KDIM    (NCELL*LEN_T) // 20480 features per batch row
#define NOUT    10
#define NPAD    16            // padded output columns for 16x16 WMMA tile

typedef _Float16 v8h  __attribute__((ext_vector_type(8)));
typedef _Float16 v16h __attribute__((ext_vector_type(16)));
typedef float    v8f  __attribute__((ext_vector_type(8)));

// =====================================================================
// Kernel 1: thermally-coupled neuristor lattice, Euler-Maruyama,
// fused rising-edge peak detection + 50-step binning -> f16 features.
// One workgroup (1024 threads = 32 wave32) per batch element.
// Double-buffered T grid in LDS -> a single barrier per time step:
// a wave can only write buffer (s%2) at step s+2 after passing the
// step s+1 barrier, which requires all waves to have finished their
// step-s reads of that buffer.
// =====================================================================
__global__ __launch_bounds__(NCELL) void sim_kernel(const float* __restrict__ x,
                                                    _Float16* __restrict__ feats) {
    const int b    = blockIdx.x;
    const int cell = threadIdx.x;
    const int ix   = cell >> 5;
    const int iy   = cell & 31;

    __shared__ float Ts[2 * NCELL];   // double-buffered 32x32 T grid (8 KB)

    // torus stencil neighbors (jnp.roll semantics)
    const int up = (((ix + 31) & 31) << 5) | iy;
    const int dn = (((ix +  1) & 31) << 5) | iy;
    const int lf = (ix << 5) | ((iy + 31) & 31);
    const int rt = (ix << 5) | ((iy +  1) & 31);

    const float V = 11.0f + 2.0f * x[b * NCELL + cell];

    float v = 0.0f;
    float T = 325.0f;

    // per-thread xorshift RNG for the 1e-3-scale thermal noise
    unsigned rng = (unsigned)(b * NCELL + cell) * 2654435761u ^ 0x9E3779B9u;
    rng ^= rng << 13; rng ^= rng >> 17; rng ^= rng << 5;

    float prevI    = 3.4e38f;   // +inf sentinel: no spike at step 0
    int   cooldown = 0;
    int   count    = 0;
    int   binCtr   = 0;
    int   binIdx   = 0;
    float zCache   = 0.0f;
    bool  haveZ    = false;

    _Float16* featRow = feats + (size_t)b * KDIM + cell * LEN_T;

    const float INV_TW   = 0.2f;          // 1 / (T_W=5)
    const float LOG_RINS = 4.6051702f;    // ln(100)
    const float NOISE_SD = 0.0031622777f; // 0.001 * sqrt(10)

    for (int step = 0; step < NSTEP; ++step) {
        float* buf = Ts + ((step & 1) << 10);
        buf[cell] = T;
        __syncthreads();
        const float lap = buf[up] + buf[dn] + buf[lf] + buf[rt] - 4.0f * T;

        const float e = __expf(-(T - 340.0f) * INV_TW);
        const float s = __builtin_amdgcn_rcpf(1.0f + e);     // fast v_rcp_f32
        const float G = __expf(-(1.0f - s) * LOG_RINS);
        const float I = v * G;

        // Box-Muller, one normal per step (pair generated every other step;
        // haveZ flips uniformly across the wave -> uniform branch)
        float z;
        if (!haveZ) {
            rng ^= rng << 13; rng ^= rng >> 17; rng ^= rng << 5;
            float u1 = (float)(rng >> 8) * 5.9604645e-8f;
            rng ^= rng << 13; rng ^= rng >> 17; rng ^= rng << 5;
            float u2 = (float)(rng >> 8) * 5.9604645e-8f;
            u1 = fmaxf(u1, 1.0e-7f);
            const float r  = sqrtf(-2.0f * __logf(u1));
            const float th = 6.2831853f * u2;
            z      = r * __cosf(th);
            zCache = r * __sinf(th);
            haveZ  = true;
        } else {
            z     = zCache;
            haveZ = false;
        }

        const float v_new = v + 0.1f  * ((V - v) * (1.0f / 12.0f) - I);
        const float T_new = T + 0.01f * (v * I - (T - 325.0f) + 0.02f * lap)
                              + NOISE_SD * z;
        v = v_new;
        T = T_new;

        // rising-edge spike with refractory MIN_DIST=101
        const bool is_peak = (I > 1.5f) && (prevI <= 1.5f) && (cooldown <= 0);
        cooldown = is_peak ? 101 : (cooldown - 1);
        prevI    = I;
        count   += is_peak ? 1 : 0;

        if (++binCtr == LEN_Y) {
            // feature index = cell*LEN_T + bin  (matches reference reshape)
            featRow[binIdx] = (_Float16)count;
            count  = 0;
            binCtr = 0;
            ++binIdx;
        }
    }
}

// =====================================================================
// Kernel 2: convert W (10 x 20480 f32) -> padded f16 B buffer (16 x 20480),
// rows 10..15 zero so the WMMA tile needs no masking.
// =====================================================================
__global__ void wconv_kernel(const float* __restrict__ W, _Float16* __restrict__ Bmat) {
    const int idx = blockIdx.x * 256 + threadIdx.x;
    if (idx >= NPAD * KDIM) return;
    const int n = idx / KDIM;
    const int k = idx - n * KDIM;
    const float val = (n < NOUT) ? W[n * KDIM + k] : 0.0f;
    Bmat[idx] = (_Float16)val;
}

// =====================================================================
// Kernel 3: logits = feats(64x20480 f16) @ Wpad^T via v_wmma_f32_16x16x32_f16.
// 4 blocks x 1 wave32; each wave owns one 16x16 output tile, K-loop of 640.
// Per-lane loads match the ISA VGPR striping:
//   A 16x32 f16 : lane<16 -> row=lane,   K = {k..k+7, k+16..k+23}
//                 lane>=16 -> row=lane-16,K = {k+8..k+15, k+24..k+31}
//   B 32x16 f16 : lane<16 -> col=lane,   K = {k..k+7, k+16..k+23}
//                 lane>=16 -> col=lane-16,K = {k+8..k+15, k+24..k+31}
// =====================================================================
__global__ __launch_bounds__(32) void gemm_kernel(const _Float16* __restrict__ feats,
                                                  const _Float16* __restrict__ Bmat,
                                                  float* __restrict__ logits) {
    const int m0   = blockIdx.x * 16;
    const int lane = threadIdx.x;
    const int hi   = lane >> 4;        // 0: lanes 0-15, 1: lanes 16-31
    const int lr   = lane & 15;

    const _Float16* aRow = feats + (size_t)(m0 + lr) * KDIM + hi * 8;
    const _Float16* bCol = Bmat  + (size_t)lr * KDIM + hi * 8;

    v8f acc = {};
    for (int k = 0; k < KDIM; k += 32) {
        __builtin_prefetch((const void*)(aRow + k + 512), 0, 1); // global_prefetch_b8
        __builtin_prefetch((const void*)(bCol + k + 512), 0, 1);

        const v8h a0 = *(const v8h*)(aRow + k);        // A: K-block low octet
        const v8h a1 = *(const v8h*)(aRow + k + 16);   // A: K-block high octet
        const v8h b0 = *(const v8h*)(bCol + k);        // B: K-block low octet
        const v8h b1 = *(const v8h*)(bCol + k + 16);   // B: K-block high octet

        v16h a, bb;
#pragma unroll
        for (int i = 0; i < 8; ++i) {
            a[i]      = a0[i];
            a[i + 8]  = a1[i];
            bb[i]     = b0[i];
            bb[i + 8] = b1[i];
        }
        acc = __builtin_amdgcn_wmma_f32_16x16x32_f16(
            /*neg_a=*/false, a, /*neg_b=*/false, bb,
            /*c_mod=*/(short)0, acc, /*reuse_a=*/false, /*reuse_b=*/false);
    }

    // C/D layout: lane col = lane&15; VGPR r -> row r (+8 for lanes 16-31)
#pragma unroll
    for (int r = 0; r < 8; ++r) {
        logits[(m0 + r + hi * 8) * NPAD + lr] = acc[r];
    }
}

// =====================================================================
// Kernel 4: bias add + log_softmax over the 10 real columns.
// =====================================================================
__global__ __launch_bounds__(BATCH) void lsm_kernel(const float* __restrict__ logits,
                                                    const float* __restrict__ bias,
                                                    float* __restrict__ out) {
    const int row = threadIdx.x;   // 64 threads, one row each
    float vals[NOUT];
    float mx = -3.4e38f;
#pragma unroll
    for (int j = 0; j < NOUT; ++j) {
        vals[j] = logits[row * NPAD + j] + bias[j];
        mx = fmaxf(mx, vals[j]);
    }
    float se = 0.0f;
#pragma unroll
    for (int j = 0; j < NOUT; ++j) se += __expf(vals[j] - mx);
    const float lse = mx + __logf(se);
#pragma unroll
    for (int j = 0; j < NOUT; ++j) out[row * NOUT + j] = vals[j] - lse;
}

// =====================================================================
extern "C" void kernel_launch(void* const* d_in, const int* in_sizes, int n_in,
                              void* d_out, int out_size, void* d_ws, size_t ws_size,
                              hipStream_t stream) {
    (void)in_sizes; (void)n_in; (void)out_size; (void)ws_size;

    const float* x    = (const float*)d_in[0];  // (64, 1024)
    const float* W    = (const float*)d_in[1];  // (10, 20480)
    const float* bias = (const float*)d_in[2];  // (10,)

    char* ws = (char*)d_ws;
    _Float16* feats  = (_Float16*)ws;                                  // 64*20480*2 = 2,621,440 B
    _Float16* Bmat   = (_Float16*)(ws + (size_t)BATCH * KDIM * 2);     // 16*20480*2 =   655,360 B
    float*    logits = (float*)(ws + (size_t)BATCH * KDIM * 2
                                   + (size_t)NPAD * KDIM * 2);         // 64*16*4    =     4,096 B

    sim_kernel<<<BATCH, NCELL, 0, stream>>>(x, feats);
    wconv_kernel<<<(NPAD * KDIM + 255) / 256, 256, 0, stream>>>(W, Bmat);
    gemm_kernel<<<BATCH / 16, 32, 0, stream>>>(feats, Bmat, logits);
    lsm_kernel<<<1, BATCH, 0, stream>>>(logits, bias, (float*)d_out);
}
